// SelfAttention_64579128262969
// MI455X (gfx1250) — matrix operand order repaired and checked by reading, not compiled, not run on hardware
//
#include <hip/hip_runtime.h>
#include <hip/hip_bf16.h>

typedef __attribute__((ext_vector_type(16))) __bf16 v16bf;
typedef __attribute__((ext_vector_type(8)))  float  v8f;
typedef int gv4i __attribute__((vector_size(16)));   // <4 x i32> pointee for async b128

// ---------- CDNA5 async global->LDS path (guarded; falls back to sync copy) ----
#if __has_builtin(__builtin_amdgcn_global_load_async_to_lds_b128) && \
    __has_builtin(__builtin_amdgcn_s_wait_asynccnt)
#define USE_ASYNC 1
#else
#define USE_ASYNC 0
#endif

__device__ inline void cp16(unsigned short* lds, const unsigned short* g) {
#if USE_ASYNC
    __builtin_amdgcn_global_load_async_to_lds_b128(
        (__attribute__((address_space(1))) gv4i*)(gv4i*)(void*)(g),
        (__attribute__((address_space(3))) gv4i*)(gv4i*)(void*)(lds),
        0, 0);
#else
    *(uint4*)lds = *(const uint4*)g;
#endif
}
template <int N> __device__ inline void wait_async() {
#if USE_ASYNC
    __builtin_amdgcn_s_wait_asynccnt(N);
#endif
}

// ---------- scalar helpers ----------
__device__ inline unsigned short f2bf(float f) {
    union { float f; unsigned int u; } c; c.f = f;
    unsigned int u = c.u;
    u += 0x7FFFu + ((u >> 16) & 1u);       // round-to-nearest-even
    return (unsigned short)(u >> 16);
}
__device__ inline float bf2f(unsigned short h) {
    union { unsigned int u; float f; } c; c.u = ((unsigned int)h) << 16;
    return c.f;
}
__device__ inline __bf16 bfbits(unsigned short h) {
    union { unsigned short u; __bf16 b; } c; c.u = h;
    return c.b;
}
// A-fragment (16-bit, 16x32) K index for vector element e, lane-half `half`
__device__ inline int afragk(int e, int half) {
    return half * 16 + e;
}

__device__ inline void store_out(unsigned short* C, size_t idx, float v) { C[idx] = f2bf(v); }
__device__ inline void store_out(float* C, size_t idx, float v)          { C[idx] = v; }

// ---------- fp32 -> bf16 convert (vectorized: float4 -> 4x bf16) ----------
__global__ void cvt_f32_bf16(const float* __restrict__ src,
                             unsigned short* __restrict__ dst, int n4) {
    int i = blockIdx.x * blockDim.x + threadIdx.x;
    int stride = gridDim.x * blockDim.x;
    for (; i < n4; i += stride) {
        float4 f = ((const float4*)src)[i];
        uint2 o;
        o.x = (unsigned)f2bf(f.x) | ((unsigned)f2bf(f.y) << 16);
        o.y = (unsigned)f2bf(f.z) | ((unsigned)f2bf(f.w) << 16);
        ((uint2*)dst)[i] = o;
    }
}

// ---------- tiled bf16 GEMM: C[M,N] = A[M,K] @ B[N,K]^T ----------
// block = 256 threads (8 waves). Block tile 128(M) x 128(N), K stage 64.
// Wave tile 32x64 -> 16 WMMA per stage per wave. Double-buffered LDS with
// async global->LDS copies pipelined one stage ahead.
template <typename OutT>
__global__ __launch_bounds__(256)
void gemm_bf16_wmma(const unsigned short* __restrict__ A,
                    const unsigned short* __restrict__ Bw,
                    OutT* __restrict__ C, int M, int N, int K) {
    __shared__ unsigned short lA[2][128][64];
    __shared__ unsigned short lB[2][128][64];
    const int t    = threadIdx.x;
    const int wave = t >> 5;
    const int lane = t & 31;
    const int n15  = lane & 15;
    const int half = lane >> 4;
    const int m0   = blockIdx.y * 128;
    const int n0   = blockIdx.x * 128;
    const int mw   = (wave & 3) * 32;   // wave M offset in block tile
    const int nw   = (wave >> 2) * 64;  // wave N offset in block tile

    v8f acc[2][4] = {};

    // stage issue: 128x64 ushorts per tile = 1024 uint4 chunks, 4 per thread
    auto stage = [&](int k0, int buf) {
        for (int c = t; c < 1024; c += 256) {
            int row = c >> 3, col = (c & 7) * 8;
            cp16(&lA[buf][row][col], &A[(size_t)(m0 + row) * K + k0 + col]);
        }
        for (int c = t; c < 1024; c += 256) {
            int row = c >> 3, col = (c & 7) * 8;
            cp16(&lB[buf][row][col], &Bw[(size_t)(n0 + row) * K + k0 + col]);
        }
    };

    const int nstage = K / 64;
    stage(0, 0);
    for (int s = 0; s < nstage; ++s) {
        const int buf = s & 1;
        if (s + 1 < nstage) {
            stage((s + 1) * 64, buf ^ 1);
            wait_async<8>();   // our 8 chunks of stage s are home; s+1 in flight
        } else {
            wait_async<0>();
        }
        __syncthreads();

        const unsigned short (*cA)[64] = lA[buf];
        const unsigned short (*cB)[64] = lB[buf];
#pragma unroll
        for (int kk = 0; kk < 64; kk += 32) {
            v16bf a[2], b[4];
#pragma unroll
            for (int mt = 0; mt < 2; ++mt)
#pragma unroll
                for (int e = 0; e < 16; ++e)
                    a[mt][e] = bfbits(cA[mw + mt * 16 + n15][kk + afragk(e, half)]);
#pragma unroll
            for (int nt = 0; nt < 4; ++nt)
#pragma unroll
                for (int e = 0; e < 16; ++e)
                    b[nt][e] = bfbits(cB[nw + nt * 16 + n15][kk + half * 16 + e]);
#pragma unroll
            for (int mt = 0; mt < 2; ++mt)
#pragma unroll
                for (int nt = 0; nt < 4; ++nt)
                    acc[mt][nt] = __builtin_amdgcn_wmma_f32_16x16x32_bf16(
                        false, a[mt], false, b[nt], (short)0, acc[mt][nt], false, false);
        }
        __syncthreads();
    }

#pragma unroll
    for (int mt = 0; mt < 2; ++mt)
#pragma unroll
        for (int nt = 0; nt < 4; ++nt)
#pragma unroll
            for (int i = 0; i < 8; ++i) {
                int m = m0 + mw + mt * 16 + half * 8 + i;
                int n = n0 + nw + nt * 16 + n15;
                store_out(C, (size_t)m * N + n, acc[mt][nt][i]);
            }
}

// ---------- per-head LayerNorm over 64 elems (in-place, bf16) ----------
__global__ __launch_bounds__(256)
void ln_head_kernel(unsigned short* __restrict__ X,
                    const float* __restrict__ w, const float* __restrict__ b,
                    int nrows) {
    int row  = blockIdx.x * 8 + (threadIdx.x >> 5);
    int lane = threadIdx.x & 31;
    if (row >= nrows) return;
    unsigned short* p = X + (size_t)row * 64;
    float x0 = bf2f(p[lane]);
    float x1 = bf2f(p[lane + 32]);
    float s = x0 + x1;
#pragma unroll
    for (int d = 16; d >= 1; d >>= 1) s += __shfl_xor(s, d, 32);
    float mu = s * (1.0f / 64.0f);
    float d0 = x0 - mu, d1 = x1 - mu;
    float vs = d0 * d0 + d1 * d1;
#pragma unroll
    for (int d = 16; d >= 1; d >>= 1) vs += __shfl_xor(vs, d, 32);
    float inv = rsqrtf(vs * (1.0f / 64.0f) + 1e-5f);
    p[lane]      = f2bf(d0 * inv * w[lane]      + b[lane]);
    p[lane + 32] = f2bf(d1 * inv * w[lane + 32] + b[lane + 32]);
}

// ---------- flash-style causal attention ----------
// grid (T/64, H, B); block 128 (4 waves, 16 query rows each). S = 64.
// K/V tiles double-buffered: async copies for tile kt+1 issued before
// computing tile kt, waiting asynccnt<=8 so tile kt's data is home.
__global__ __launch_bounds__(128)
void attn_kernel(const unsigned short* __restrict__ Q,
                 const unsigned short* __restrict__ Kb,
                 const unsigned short* __restrict__ V,
                 unsigned short* __restrict__ O,
                 int B, int T, int E) {
    const int S = 64;
    __shared__ unsigned short lq[64][64];
    __shared__ unsigned short lk[2][64][64];
    __shared__ unsigned short lv[2][64][64];
    __shared__ unsigned short lp[64][64];

    const int t    = threadIdx.x;
    const int wave = t >> 5;
    const int lane = t & 31;
    const int n15  = lane & 15;
    const int half = lane >> 4;
    const int qt = blockIdx.x, h = blockIdx.y, b = blockIdx.z;
    const int qbase = qt * 64;

    // issue stage for K/V tile kt into buffer buf: 8 uint4 chunks per thread
    auto stage_kv = [&](int kt, int buf) {
        const unsigned short* ksrc = Kb + ((size_t)b * T + kt * 64) * E + h * S;
        const unsigned short* vsrc = V  + ((size_t)b * T + kt * 64) * E + h * S;
        for (int c = t; c < 512; c += 128) {
            int r = c >> 3, col = (c & 7) * 8;
            cp16(&lk[buf][r][col], &ksrc[(size_t)r * E + col]);
            cp16(&lv[buf][r][col], &vsrc[(size_t)r * E + col]);
        }
    };

    // overlap Q-tile load with first K/V tile
    const unsigned short* qsrc = Q + ((size_t)b * T + qbase) * E + h * S;
    for (int c = t; c < 512; c += 128) {
        int r = c >> 3, col = (c & 7) * 8;
        cp16(&lq[r][col], &qsrc[(size_t)r * E + col]);
    }
    stage_kv(0, 0);
    wait_async<8>();   // our 4 Q chunks are home; KV tile 0 may be in flight
    __syncthreads();

    v16bf aq[2];
#pragma unroll
    for (int s = 0; s < 2; ++s)
#pragma unroll
        for (int e = 0; e < 16; ++e)
            aq[s][e] = bfbits(lq[wave * 16 + n15][s * 32 + afragk(e, half)]);

    v8f acco[4] = {};
    float mrow[8], lrow[8];
#pragma unroll
    for (int i = 0; i < 8; ++i) { mrow[i] = -3.0e38f; lrow[i] = 0.0f; }

    const float scale = 0.125f;  // 1/sqrt(64)

    for (int kt = 0; kt <= qt; ++kt) {
        const int buf = kt & 1;
        if (kt < qt) {
            stage_kv(kt + 1, buf ^ 1);
            wait_async<8>();   // tile kt home; kt+1 in flight
        } else {
            wait_async<0>();
        }
        __syncthreads();

        const unsigned short (*ck)[64] = lk[buf];
        const unsigned short (*cv)[64] = lv[buf];

        // scores: S = Q . K^T  (16 query rows x 64 keys per wave)
        v8f sc[4] = {};
#pragma unroll
        for (int nt = 0; nt < 4; ++nt)
#pragma unroll
            for (int s = 0; s < 2; ++s) {
                v16bf bk;
#pragma unroll
                for (int e = 0; e < 16; ++e)
                    bk[e] = bfbits(ck[nt * 16 + n15][s * 32 + half * 16 + e]);
                sc[nt] = __builtin_amdgcn_wmma_f32_16x16x32_bf16(
                    false, aq[s], false, bk, (short)0, sc[nt], false, false);
            }

        // scale + causal mask + running max
        float pm[4][8];
        float rm[8];
#pragma unroll
        for (int i = 0; i < 8; ++i) rm[i] = -3.0e38f;
#pragma unroll
        for (int nt = 0; nt < 4; ++nt)
#pragma unroll
            for (int i = 0; i < 8; ++i) {
                float v = sc[nt][i] * scale;
                if (kt == qt) {
                    int ig = wave * 16 + half * 8 + i;  // query in tile
                    int jg = nt * 16 + n15;             // key in tile
                    if (jg > ig) v = -3.0e38f;
                }
                pm[nt][i] = v;
                rm[i] = fmaxf(rm[i], v);
            }
#pragma unroll
        for (int i = 0; i < 8; ++i)
#pragma unroll
            for (int d = 1; d < 16; d <<= 1)
                rm[i] = fmaxf(rm[i], __shfl_xor(rm[i], d, 32));

        float alpha[8], rs[8];
#pragma unroll
        for (int i = 0; i < 8; ++i) {
            float mn = fmaxf(mrow[i], rm[i]);
            alpha[i] = __expf(mrow[i] - mn);
            mrow[i]  = mn;
            rs[i]    = 0.0f;
        }
#pragma unroll
        for (int nt = 0; nt < 4; ++nt)
#pragma unroll
            for (int i = 0; i < 8; ++i) {
                float p = __expf(pm[nt][i] - mrow[i]);
                pm[nt][i] = p;
                rs[i] += p;
            }
#pragma unroll
        for (int i = 0; i < 8; ++i)
#pragma unroll
            for (int d = 1; d < 16; d <<= 1)
                rs[i] += __shfl_xor(rs[i], d, 32);
#pragma unroll
        for (int i = 0; i < 8; ++i) lrow[i] = lrow[i] * alpha[i] + rs[i];
#pragma unroll
        for (int nt = 0; nt < 4; ++nt)
#pragma unroll
            for (int i = 0; i < 8; ++i) acco[nt][i] *= alpha[i];

        // stage P (C-layout -> LDS) and reload as A-fragment (wave-local)
#pragma unroll
        for (int nt = 0; nt < 4; ++nt)
#pragma unroll
            for (int i = 0; i < 8; ++i)
                lp[wave * 16 + half * 8 + i][nt * 16 + n15] = f2bf(pm[nt][i]);

        // O += P . V
#pragma unroll
        for (int nt = 0; nt < 4; ++nt)  // output head-dim tile
#pragma unroll
            for (int s = 0; s < 2; ++s) {
                v16bf ap, bv;
#pragma unroll
                for (int e = 0; e < 16; ++e) {
                    ap[e] = bfbits(lp[wave * 16 + n15][s * 32 + afragk(e, half)]);
                    bv[e] = bfbits(cv[s * 32 + half * 16 + e][nt * 16 + n15]);
                }
                acco[nt] = __builtin_amdgcn_wmma_f32_16x16x32_bf16(
                    false, ap, false, bv, (short)0, acco[nt], false, false);
            }
        __syncthreads();  // all readers done before next tile overwrites buffers
    }

    // normalize and store attention output (bf16) for the final projection
    unsigned short* odst = O + ((size_t)b * T + qbase + wave * 16) * E + h * S;
#pragma unroll
    for (int nt = 0; nt < 4; ++nt)
#pragma unroll
        for (int i = 0; i < 8; ++i) {
            float v = acco[nt][i] / lrow[i];
            odst[(size_t)(half * 8 + i) * E + nt * 16 + n15] = f2bf(v);
        }
}

// ---------- host launch ----------
extern "C" void kernel_launch(void* const* d_in, const int* in_sizes, int n_in,
                              void* d_out, int out_size, void* d_ws, size_t ws_size,
                              hipStream_t stream) {
    const int B = 4, T = 2048, E = 1024, H = 16;
    const int M = B * T;

    const float* x    = (const float*)d_in[0];
    const float* Wk   = (const float*)d_in[1];
    const float* Wq   = (const float*)d_in[2];
    const float* Wv   = (const float*)d_in[3];
    const float* Wu   = (const float*)d_in[4];
    const float* klnw = (const float*)d_in[5];
    const float* klnb = (const float*)d_in[6];
    const float* qlnw = (const float*)d_in[7];
    const float* qlnb = (const float*)d_in[8];

    char* ws = (char*)d_ws;
    size_t off = 0;
    auto alloc = [&](size_t bytes) -> void* {
        void* p = ws + off;
        off = (off + bytes + 255) & ~(size_t)255;
        return p;
    };
    unsigned short* xb  = (unsigned short*)alloc((size_t)M * E * 2);
    unsigned short* wkb = (unsigned short*)alloc((size_t)E * E * 2);
    unsigned short* wqb = (unsigned short*)alloc((size_t)E * E * 2);
    unsigned short* wvb = (unsigned short*)alloc((size_t)E * E * 2);
    unsigned short* wub = (unsigned short*)alloc((size_t)E * E * 2);
    unsigned short* qb  = (unsigned short*)alloc((size_t)M * E * 2);
    unsigned short* kb  = (unsigned short*)alloc((size_t)M * E * 2);
    unsigned short* vb  = (unsigned short*)alloc((size_t)M * E * 2);
    unsigned short* ab  = (unsigned short*)alloc((size_t)M * E * 2);

    cvt_f32_bf16<<<2048, 256, 0, stream>>>(x,  xb,  (M * E) / 4);
    cvt_f32_bf16<<<512,  256, 0, stream>>>(Wk, wkb, (E * E) / 4);
    cvt_f32_bf16<<<512,  256, 0, stream>>>(Wq, wqb, (E * E) / 4);
    cvt_f32_bf16<<<512,  256, 0, stream>>>(Wv, wvb, (E * E) / 4);
    cvt_f32_bf16<<<512,  256, 0, stream>>>(Wu, wub, (E * E) / 4);

    dim3 ggrid(E / 128, M / 128);
    gemm_bf16_wmma<<<ggrid, 256, 0, stream>>>(xb, wqb, qb, M, E, E);
    gemm_bf16_wmma<<<ggrid, 256, 0, stream>>>(xb, wkb, kb, M, E, E);
    gemm_bf16_wmma<<<ggrid, 256, 0, stream>>>(xb, wvb, vb, M, E, E);

    ln_head_kernel<<<(M * H) / 8, 256, 0, stream>>>(kb, klnw, klnb, M * H);
    ln_head_kernel<<<(M * H) / 8, 256, 0, stream>>>(qb, qlnw, qlnb, M * H);

    attn_kernel<<<dim3(T / 64, H, B), 128, 0, stream>>>(qb, kb, vb, ab, B, T, E);

    gemm_bf16_wmma<<<ggrid, 256, 0, stream>>>(ab, wub, (float*)d_out, M, E, E);
}
